// Diffuser_ped_inter_geometric_cond_w_history_37314675868554
// MI455X (gfx1250) — matrix operand order repaired and checked by reading, not compiled, not run on hardware
//
#include <hip/hip_runtime.h>
#include <hip/hip_bf16.h>
#include <math.h>

// ---------------- problem constants ----------------
#define BB 32
#define NN 512
#define KK 16
#define TT 8
#define HID 64
#define SS 32
#define EHH 32
#define HLL 32
#define OLL 16
#define NNODES (BB * NN)   // 16384

typedef __attribute__((ext_vector_type(16))) _Float16 v16h;
typedef __attribute__((ext_vector_type(8)))  float    v8f;

// ---------------- WMMA helpers (CDNA5 wave32 layouts, ISA 7.12.2) ----------------
__device__ __forceinline__ v8f wmma_f16(v16h a, v16h b, v8f c) {
  // D = A(16x32 f16) * B(32x16 f16) + C(16x16 f32)
  return __builtin_amdgcn_wmma_f32_16x16x32_f16(false, a, false, b, (short)0, c,
                                                false, false);
}

// A-matrix 16x32 f16 from LDS, row-major with `strideH` halfs per row.
// Lane L: M = L%16 ; lanes 0-15 hold K=0..15, lanes 16-31 K=16..31.
__device__ __forceinline__ v16h load_A16(const _Float16* p, int strideH) {
  const int lane = threadIdx.x & 31;
  const int m  = lane & 15;
  const int kb = (lane < 16) ? 0 : 16;
  const _Float16* row = p + m * strideH + kb;
  v16h a;
#pragma unroll
  for (int i = 0; i < 8; ++i) a[i] = row[i];
#pragma unroll
  for (int i = 0; i < 8; ++i) a[8 + i] = row[8 + i];
  return a;
}

// B-matrix tile from a pre-swizzled global buffer: [tile][lane][16 halfs].
// Each lane reads 32 contiguous bytes -> 2x global_load_b128.
__device__ __forceinline__ v16h load_B_swz(const _Float16* base, int tix) {
  const int lane = threadIdx.x & 31;
  const _Float16* p = base + ((size_t)tix * 32 + lane) * 16;
  v16h b;
#pragma unroll
  for (int i = 0; i < 16; ++i) b[i] = p[i];
  return b;
}

// Fast transcendentals: v_exp_f32 directly, no libm slow path / divergence.
// Pre-activations here are O(1), so no range-reduction needed.
__device__ __forceinline__ float fexp_(float x)     { return __expf(x); }
__device__ __forceinline__ float sigmoidf_(float x) { return 1.0f / (1.0f + fexp_(-x)); }
__device__ __forceinline__ float tanhf_(float x)    { return 1.0f - 2.0f / (fexp_(2.0f * x) + 1.0f); }
__device__ __forceinline__ float siluf_(float x)    { return x * sigmoidf_(x); }
__device__ __forceinline__ float reluf_(float x)    { return x > 0.0f ? x : 0.0f; }

// ---------------- workspace layout (bytes) ----------------
// Swizzled B buffers: tile = 32 lanes x 16 halfs = 512 halfs = 1024 B.
#define WS_WIHZ   ((size_t)0)           // 8 tiles  = 8192 B
#define WS_WHHZ   ((size_t)8192)        // 8 tiles  = 8192 B
#define WS_WLOZ   ((size_t)16384)       // 1 tile   = 1024 B
#define WS_WE1Z   ((size_t)17408)       // 2 layers x 16 tiles = 32768 B
#define WS_WE2Z   ((size_t)50176)       // 2 layers x 8 tiles  = 16384 B
#define WS_WHZ    ((size_t)66560)       // 2 layers x 16 tiles = 32768 B
#define WS_WLAST  ((size_t)99328)       // 2*64 f32 = 512 B
#define WS_BG     ((size_t)99840)       // 128 f32  = 512 B
#define WS_HA     ((size_t)100352)      // 16384*64 f32 = 4 MB
#define WS_HB     ((size_t)4294656)
#define WS_HISTE  ((size_t)8488960)     // 16384*16 f32
#define WS_SP     ((size_t)9537536)     // 16384*32 f32
#define WS_TIME   ((size_t)11634688)    // 16384*3  f32
#define WS_PACC   ((size_t)11831296)    // 16384*2  f32
#define WS_COORD  ((size_t)11962368)    // 16384*2  f32

// ---------------- kernel 1: swizzle weights into WMMA B fragment layout ----------
// src: original (N_out rows, K_in cols) f32 row-major, row stride = kstride.
// dst[(tix*32 + lane)*16 + i] = src[n][k]:
//   nt = tix/KS, ks = tix%KS, n = (lane&15)+16*nt, k = 32*ks + (lane<16?0:16) + i
__device__ void fill_swz(const float* src, int kstride, int NT, int KS,
                         _Float16* dst, int gid, int str) {
  const int total = NT * KS * 512;
  for (int idx = gid; idx < total; idx += str) {
    int tix = idx >> 9;
    int rem = idx & 511;
    int l = rem >> 4;
    int i = rem & 15;
    int nt = tix / KS, ks = tix % KS;
    int n = (l & 15) + nt * 16;
    int k = ks * 32 + ((l < 16) ? 0 : 16) + i;
    dst[idx] = (_Float16)src[(size_t)n * kstride + k];
  }
}

__global__ void prep_weights(const float* We1, const float* We2, const float* Wh,
                             const float* Wih, const float* Whh, const float* Wlo,
                             const float* bih, const float* bhh,
                             _Float16* wihz, _Float16* whhz, _Float16* wloz,
                             _Float16* we1z, _Float16* we2z, _Float16* whz,
                             float* wlast, float* bg) {
  const int gid = blockIdx.x * blockDim.x + threadIdx.x;
  const int str = gridDim.x * blockDim.x;
  fill_swz(Wih, 32, 8, 1, wihz, gid, str);                 // gates: N=128, K=32
  fill_swz(Whh, 32, 8, 1, whhz, gid, str);
  fill_swz(Wlo, 32, 1, 1, wloz, gid, str);                 // N=16,  K=32
  for (int l = 0; l < 2; ++l) {
    fill_swz(We1 + (size_t)l * 64 * 129, 129, 4, 4, we1z + (size_t)l * 16 * 512, gid, str);
    fill_swz(We2 + (size_t)l * 64 * 64,   64, 4, 2, we2z + (size_t)l * 8 * 512,  gid, str);
    fill_swz(Wh  + (size_t)l * 64 * 128, 128, 4, 4, whz  + (size_t)l * 16 * 512, gid, str);
  }
  for (int i = gid; i < 2 * 64; i += str)                  // last column of We1 (d^2)
    wlast[i] = We1[((size_t)i) * 129 + 128];
  for (int i = gid; i < 128; i += str) bg[i] = bih[i] + bhh[i];
}

// ---------------- kernel 2: per-node prep (coords, time, sp, pacc, embed) ----------
__global__ void prep_nodes(const float* x, const float* beta, const float* ped,
                           const float* selff, const float* Wsp, const float* bsp,
                           const float* eW, const float* eb,
                           float* coord, float* timee, float* sp, float* pacc,
                           float* h0) {
  const int i = blockIdx.x * blockDim.x + threadIdx.x;
  if (i >= NNODES) return;
  const int b = i >> 9;                         // N = 512
  const float* pf = ped + (size_t)i * 4;
  coord[i * 2 + 0] = pf[0];
  coord[i * 2 + 1] = pf[1];
  const float bv = beta[b];
  const float te0 = bv, te1 = __sinf(bv), te2 = __cosf(bv);
  timee[i * 3 + 0] = te0; timee[i * 3 + 1] = te1; timee[i * 3 + 2] = te2;
  const float* xv = x + (size_t)i * 2;
#pragma unroll
  for (int o = 0; o < SS; ++o)
    sp[(size_t)i * SS + o] = reluf_(bsp[o] + Wsp[o * 2 + 0] * xv[0] + Wsp[o * 2 + 1] * xv[1]);
  const float* sf = selff + (size_t)i * 5;
  float nrm = sqrtf(sf[0] * sf[0] + sf[1] * sf[1]);
  float den = (nrm == 0.0f) ? (nrm + 0.1f) : nrm;
  float dx = sf[0] / den, dy = sf[1] / den;
  pacc[i * 2 + 0] = (sf[4] * dx - sf[2]) * 0.5f;   // /TAU, TAU=2
  pacc[i * 2 + 1] = (sf[4] * dy - sf[3]) * 0.5f;
  float in7[7] = {pf[0], pf[1], pf[2], pf[3], te0, te1, te2};
#pragma unroll
  for (int o = 0; o < HID; ++o) {
    float acc = eb[o];
#pragma unroll
    for (int c = 0; c < 7; ++c) acc += eW[o * 7 + c] * in7[c];
    h0[(size_t)i * HID + o] = siluf_(acc);
  }
}

// ---------------- kernel 3: history embed + LSTM + out projection ----------------
// 1 wave = 16 (b,n) rows. Gates: M=16 rows, K=32, N=128 -> 16 WMMA / step.
// All 16 gate-weight B tiles are loop-invariant: preloaded into registers.
__global__ void __launch_bounds__(128)
lstm_kernel(const float* hist, const float* Whist, const float* bhist,
            const _Float16* wihz, const _Float16* whhz, const float* bg,
            const _Float16* wloz, const float* blo, float* hist_emb) {
  __shared__ _Float16 lds_he[4][16 * TT * EHH];   // [wave][row*256 + t*32 + e]
  __shared__ _Float16 lds_h[4][16 * HLL];         // current h, row-major 16x32
  const int wave = threadIdx.x >> 5;
  const int lane = threadIdx.x & 31;
  const int hi8  = (lane >> 4) << 3;
  const int r0   = (blockIdx.x * 4 + wave) * 16;  // row base, 1024 tiles total

  // history embed: he = relu(hist @ Whist^T + bhist), 16 rows x 8 steps x 32
  for (int o = lane; o < 16 * TT * EHH; o += 32) {
    int row = o >> 8, t = (o >> 5) & 7, e = o & 31;
    const float* hx = hist + ((size_t)(r0 + row) * TT + t) * 2;
    float v = bhist[e] + Whist[e * 2 + 0] * hx[0] + Whist[e * 2 + 1] * hx[1];
    lds_he[wave][o] = (_Float16)reluf_(v);
  }
  for (int o = lane; o < 16 * HLL; o += 32) lds_h[wave][o] = (_Float16)0.0f;

  // preload loop-invariant B tiles + gate biases
  v16h Bih[8], Bhh[8];
  float bgv[8];
#pragma unroll
  for (int nt = 0; nt < 8; ++nt) {
    Bih[nt] = load_B_swz(wihz, nt);
    Bhh[nt] = load_B_swz(whhz, nt);
    bgv[nt] = bg[(lane & 15) + nt * 16];
  }
  float cst[2][8];
#pragma unroll
  for (int s = 0; s < 2; ++s)
#pragma unroll
    for (int j = 0; j < 8; ++j) cst[s][j] = 0.0f;
  __syncthreads();

  for (int t = 0; t < TT; ++t) {
    v16h ax = load_A16(&lds_he[wave][t * 32], 256);
    v16h ah = load_A16(lds_h[wave], 32);
    v8f g[8];
#pragma unroll
    for (int nt = 0; nt < 8; ++nt) {
      v8f c;
#pragma unroll
      for (int j = 0; j < 8; ++j) c[j] = bgv[nt];
      c = wmma_f16(ax, Bih[nt], c);
      c = wmma_f16(ah, Bhh[nt], c);
      g[nt] = c;
    }
    __syncthreads();   // done reading previous lds_h
#pragma unroll
    for (int s = 0; s < 2; ++s) {
#pragma unroll
      for (int j = 0; j < 8; ++j) {
        float iv = sigmoidf_(g[0 + s][j]);
        float fv = sigmoidf_(g[2 + s][j]);
        float gv = tanhf_(g[4 + s][j]);
        float ov = sigmoidf_(g[6 + s][j]);
        float cn = fv * cst[s][j] + iv * gv;
        cst[s][j] = cn;
        float hn = ov * tanhf_(cn);
        lds_h[wave][(j + hi8) * 32 + (lane & 15) + s * 16] = (_Float16)hn;
      }
    }
    __syncthreads();
  }

  // hist_embedded = hT @ Wlo^T + blo  (M=16, K=32, N=16)
  v16h ah = load_A16(lds_h[wave], 32);
  v8f c;
  float bv = blo[lane & 15];
#pragma unroll
  for (int j = 0; j < 8; ++j) c[j] = bv;
  c = wmma_f16(ah, load_B_swz(wloz, 0), c);
#pragma unroll
  for (int j = 0; j < 8; ++j)
    hist_emb[(size_t)(r0 + j + hi8) * OLL + (lane & 15)] = c[j];
}

// ---------------- kernel 4: one EGNN layer ----------------
// Block = 16 nodes (same batch element), 8 waves. Each wave: 2 nodes' edge MLPs.
__global__ void __launch_bounds__(256)
egnn_layer(const float* hin, float* hout,
           const _Float16* we1z, const float* wlast, const float* b1,
           const _Float16* we2z, const float* b2,
           const _Float16* whz, const float* bh,
           const float* coord, const int* nidx, const int* nmask) {
  __shared__ _Float16 hi_l[16 * 64];        // node h (f16)
  __shared__ _Float16 acat[16 * 128];       // [h | agg] A-matrix for node update
  __shared__ _Float16 ebuf[8][16 * 128];    // per-wave edge A-matrix
  __shared__ _Float16 actb[8][16 * 64];     // per-wave Lin1 activations
  __shared__ float    d2b[8][16];
  __shared__ float    mkb[8][16];
  const int wave = threadIdx.x >> 5;
  const int lane = threadIdx.x & 31;
  const int hi8  = (lane >> 4) << 3;
  const int g0   = blockIdx.x * 16;          // node base
  const int bb   = g0 >> 9;                  // batch element

  for (int o = threadIdx.x; o < 16 * 64; o += 256) {
    _Float16 hv = (_Float16)hin[(size_t)g0 * 64 + o];
    hi_l[o] = hv;
    acat[(o >> 6) * 128 + (o & 63)] = hv;
  }
  __syncthreads();

  for (int rep = 0; rep < 2; ++rep) {
    const int nodeL = wave * 2 + rep;
    const int gnode = g0 + nodeL;
    // build edge features: row k = [ h_i(64) | h_nb(64) ], plus d2 & mask
    {
      const int k   = lane & 15;
      const int nb  = nidx[(size_t)gnode * KK + k];
      const int gnb = (bb << 9) + nb;
      if (lane < 16) {
#pragma unroll
        for (int c = 0; c < 64; ++c)
          ebuf[wave][k * 128 + c] = hi_l[nodeL * 64 + c];
        float dx = coord[(size_t)gnode * 2 + 0] - coord[(size_t)gnb * 2 + 0];
        float dy = coord[(size_t)gnode * 2 + 1] - coord[(size_t)gnb * 2 + 1];
        d2b[wave][k] = dx * dx + dy * dy;
        mkb[wave][k] = (float)nmask[(size_t)gnode * KK + k];
      } else {
        const float* hn = hin + (size_t)gnb * 64;
#pragma unroll
        for (int c = 0; c < 64; ++c)
          ebuf[wave][k * 128 + 64 + c] = (_Float16)hn[c];
      }
    }
    __syncthreads();

    // Lin1: M=16 edges, K=128 (+ d2 column folded into C init), N=64, then SiLU
#pragma unroll
    for (int nt = 0; nt < 4; ++nt) {
      const int n = (lane & 15) + nt * 16;
      const float bv = b1[n], wl = wlast[n];
      v8f c;
#pragma unroll
      for (int j = 0; j < 8; ++j) c[j] = bv + d2b[wave][j + hi8] * wl;
#pragma unroll
      for (int ks = 0; ks < 4; ++ks)
        c = wmma_f16(load_A16(&ebuf[wave][ks * 32], 128),
                     load_B_swz(we1z, nt * 4 + ks), c);
#pragma unroll
      for (int j = 0; j < 8; ++j)
        actb[wave][(j + hi8) * 64 + n] = (_Float16)siluf_(c[j]);
    }
    __syncthreads();

    // Lin2 + SiLU + mask + K-reduction -> agg row of acat
#pragma unroll
    for (int nt = 0; nt < 4; ++nt) {
      const int n = (lane & 15) + nt * 16;
      v8f c;
      const float bv = b2[n];
#pragma unroll
      for (int j = 0; j < 8; ++j) c[j] = bv;
#pragma unroll
      for (int ks = 0; ks < 2; ++ks)
        c = wmma_f16(load_A16(&actb[wave][ks * 32], 64),
                     load_B_swz(we2z, nt * 2 + ks), c);
      float part = 0.0f;
#pragma unroll
      for (int j = 0; j < 8; ++j)
        part += siluf_(c[j]) * mkb[wave][j + hi8];
      part += __shfl_xor(part, 16, 32);        // lanes L and L+16 share N
      if (lane < 16) acat[nodeL * 128 + 64 + n] = (_Float16)part;
    }
    __syncthreads();
  }

  // node update: h += [h|agg] @ Wh^T + bh   (M=16 nodes, K=128, N=64)
  if (wave < 4) {
    const int nt = wave;
    const int n = (lane & 15) + nt * 16;
    v8f c;
    const float bv = bh[n];
#pragma unroll
    for (int j = 0; j < 8; ++j) c[j] = bv;
#pragma unroll
    for (int ks = 0; ks < 4; ++ks)
      c = wmma_f16(load_A16(&acat[ks * 32], 128), load_B_swz(whz, nt * 4 + ks), c);
#pragma unroll
    for (int j = 0; j < 8; ++j) {
      const int M = j + hi8;
      hout[(size_t)(g0 + M) * 64 + n] = hin[(size_t)(g0 + M) * 64 + n] + c[j];
    }
  }
}

// ---------------- kernel 5: final concat + decode ----------------
__global__ void final_decode(const float* hA, const float* hist_emb, const float* sp,
                             const float* timee, const float* pacc,
                             const float* Wc, const float* bc,
                             const float* Wd, const float* bd, float* out) {
  const int i = blockIdx.x * blockDim.x + threadIdx.x;
  if (i >= NNODES) return;
  const float* h  = hA + (size_t)i * 64;
  const float* he = hist_emb + (size_t)i * 16;
  const float* s  = sp + (size_t)i * 32;
  const float* te = timee + (size_t)i * 3;
  float z[16];
#pragma unroll
  for (int o = 0; o < 16; ++o) {
    const float* w = Wc + o * 115;
    float acc = bc[o];
#pragma unroll
    for (int c = 0; c < 64; ++c) acc += w[c] * h[c];
#pragma unroll
    for (int c = 0; c < 16; ++c) acc += w[64 + c] * he[c];
#pragma unroll
    for (int c = 0; c < 32; ++c) acc += w[80 + c] * s[c];
#pragma unroll
    for (int c = 0; c < 3; ++c) acc += w[112 + c] * te[c];
    z[o] = reluf_(acc);
  }
#pragma unroll
  for (int d = 0; d < 2; ++d) {
    float acc = bd[d];
#pragma unroll
    for (int c = 0; c < 16; ++c) acc += Wd[d * 16 + c] * z[c];
    out[(size_t)i * 2 + d] = acc + pacc[i * 2 + d];
  }
}

// ---------------- launch ----------------
extern "C" void kernel_launch(void* const* d_in, const int* in_sizes, int n_in,
                              void* d_out, int out_size, void* d_ws, size_t ws_size,
                              hipStream_t stream) {
  const float* x       = (const float*)d_in[0];
  const float* beta    = (const float*)d_in[1];
  const float* ped     = (const float*)d_in[2];
  const float* selff   = (const float*)d_in[3];
  const float* hist    = (const float*)d_in[4];
  const float* Wsp     = (const float*)d_in[5];
  const float* bsp     = (const float*)d_in[6];
  const float* Whist   = (const float*)d_in[7];
  const float* bhist   = (const float*)d_in[8];
  const float* Wih     = (const float*)d_in[9];
  const float* Whh     = (const float*)d_in[10];
  const float* bih     = (const float*)d_in[11];
  const float* bhh     = (const float*)d_in[12];
  const float* Wlo     = (const float*)d_in[13];
  const float* blo     = (const float*)d_in[14];
  const float* Wc      = (const float*)d_in[15];
  const float* bc      = (const float*)d_in[16];
  const float* Wd      = (const float*)d_in[17];
  const float* bd      = (const float*)d_in[18];
  const float* eW      = (const float*)d_in[19];
  const float* eb      = (const float*)d_in[20];
  const float* We1     = (const float*)d_in[21];
  const float* b1      = (const float*)d_in[22];
  const float* We2     = (const float*)d_in[23];
  const float* b2      = (const float*)d_in[24];
  const float* Wh      = (const float*)d_in[25];
  const float* bh      = (const float*)d_in[26];
  const int*   nmask   = (const int*)d_in[27];
  const int*   nidx    = (const int*)d_in[28];
  float* out = (float*)d_out;
  char*  ws  = (char*)d_ws;

  _Float16* wihz  = (_Float16*)(ws + WS_WIHZ);
  _Float16* whhz  = (_Float16*)(ws + WS_WHHZ);
  _Float16* wloz  = (_Float16*)(ws + WS_WLOZ);
  _Float16* we1z  = (_Float16*)(ws + WS_WE1Z);
  _Float16* we2z  = (_Float16*)(ws + WS_WE2Z);
  _Float16* whz   = (_Float16*)(ws + WS_WHZ);
  float*    wlast = (float*)   (ws + WS_WLAST);
  float*    bg    = (float*)   (ws + WS_BG);
  float*    hA    = (float*)   (ws + WS_HA);
  float*    hB    = (float*)   (ws + WS_HB);
  float*    histe = (float*)   (ws + WS_HISTE);
  float*    sp    = (float*)   (ws + WS_SP);
  float*    timee = (float*)   (ws + WS_TIME);
  float*    pacc  = (float*)   (ws + WS_PACC);
  float*    coord = (float*)   (ws + WS_COORD);

  prep_weights<<<64, 256, 0, stream>>>(We1, We2, Wh, Wih, Whh, Wlo, bih, bhh,
                                       wihz, whhz, wloz, we1z, we2z, whz, wlast, bg);
  prep_nodes<<<NNODES / 256, 256, 0, stream>>>(x, beta, ped, selff, Wsp, bsp, eW, eb,
                                               coord, timee, sp, pacc, hA);
  lstm_kernel<<<NNODES / (16 * 4), 128, 0, stream>>>(hist, Whist, bhist, wihz, whhz,
                                                     bg, wloz, blo, histe);
  // EGNN layer 0: hA -> hB
  egnn_layer<<<NNODES / 16, 256, 0, stream>>>(hA, hB,
      we1z, wlast, b1, we2z, b2, whz, bh, coord, nidx, nmask);
  // EGNN layer 1: hB -> hA
  egnn_layer<<<NNODES / 16, 256, 0, stream>>>(hB, hA,
      we1z + 16 * 512, wlast + 64, b1 + 64,
      we2z + 8 * 512,  b2 + 64,
      whz + 16 * 512,  bh + 64, coord, nidx, nmask);
  final_decode<<<NNODES / 256, 256, 0, stream>>>(hA, histe, sp, timee, pacc,
                                                 Wc, bc, Wd, bd, out);
  (void)in_sizes; (void)n_in; (void)out_size; (void)ws_size;
}
